// LocalAttn_54614804136686
// MI455X (gfx1250) — matrix operand order repaired and checked by reading, compiled, not run on hardware
//
#include <hip/hip_runtime.h>
#include <hip/hip_bf16.h>
#include <stdint.h>

typedef __bf16 bf16;
typedef __bf16 v16bf __attribute__((ext_vector_type(16)));
typedef float  v8f   __attribute__((ext_vector_type(8)));

union BFrag { v16bf v; uint4 q[2]; };

#define WMMA_BF16(A, B, C) \
  __builtin_amdgcn_wmma_f32_16x16x32_bf16(false, (A), false, (B), (short)0, (C), false, false)

static __device__ __forceinline__ void zero_frag(BFrag& f) {
  f.q[0] = make_uint4(0u, 0u, 0u, 0u);
  f.q[1] = make_uint4(0u, 0u, 0u, 0u);
}

// Build an A-fragment (16x32, contiguous k order used by this file) from an fp32 weight row.
// lanes 0-15 take K = c0..c0+15; lanes 16-31 (caller passes
// c0 already offset by +16) take K = 16..31.
static __device__ __forceinline__ v16bf pack_a(const float* __restrict__ Wrow, int c0) {
  float4 w0 = *(const float4*)(Wrow + c0);
  float4 w1 = *(const float4*)(Wrow + c0 + 4);
  float4 w2 = *(const float4*)(Wrow + c0 + 8);
  float4 w3 = *(const float4*)(Wrow + c0 + 12);
  v16bf a;
  a[0] = (bf16)w0.x;  a[1] = (bf16)w0.y;  a[2]  = (bf16)w0.z;  a[3]  = (bf16)w0.w;
  a[4] = (bf16)w1.x;  a[5] = (bf16)w1.y;  a[6]  = (bf16)w1.z;  a[7]  = (bf16)w1.w;
  a[8] = (bf16)w2.x;  a[9] = (bf16)w2.y;  a[10] = (bf16)w2.z;  a[11] = (bf16)w2.w;
  a[12] = (bf16)w3.x; a[13] = (bf16)w3.y; a[14] = (bf16)w3.z;  a[15] = (bf16)w3.w;
  return a;
}

// ---------------------------------------------------------------------------
// Kernel 1: fused projection GEMM.  Y[b][o][l] = W[o,:].x[:,l] + bias,
// o in [0,384): 0..63 theta (Wt), 64..127 phi-raw (Wp), 128..383 g-raw (Wg).
// Per block: one batch, 32 pixels; x tile staged in LDS as bf16 [l][c].
// ---------------------------------------------------------------------------
__global__ __launch_bounds__(256) void k_proj(
    const float* __restrict__ x,
    const float* __restrict__ Wt, const float* __restrict__ bt,
    const float* __restrict__ Wp, const float* __restrict__ bp,
    const float* __restrict__ Wg, const float* __restrict__ bg,
    bf16* __restrict__ Y)
{
  constexpr int LB = 32, CPAD = 528;           // 528*2B row stride, 16B aligned
  __shared__ __align__(16) bf16 xs[LB * CPAD];
  const int b  = blockIdx.x >> 7;              // 128 l-blocks per batch
  const int l0 = (blockIdx.x & 127) * LB;
  const int tid = threadIdx.x;
  const float* xb = x + (size_t)b * 512 * 4096;
  for (int idx = tid; idx < 512 * LB; idx += 256) {
    const int c = idx >> 5, li = idx & 31;     // coalesced along l
    xs[li * CPAD + c] = (bf16)xb[(size_t)c * 4096 + l0 + li];
  }
  __syncthreads();
  const int wv = tid >> 5, lane = tid & 31;
  const int col = lane & 15, hi = lane >> 4;
  for (int t = wv; t < 48; t += 8) {           // 24 o-tiles * 2 l-tiles
    const int ot = t >> 1, lt = t & 1;
    const int o0 = ot * 16;
    const float* W; const float* bias; int wo;
    if (o0 < 64)       { W = Wt; bias = bt; wo = o0; }
    else if (o0 < 128) { W = Wp; bias = bp; wo = o0 - 64; }
    else               { W = Wg; bias = bg; wo = o0 - 128; }
    v8f acc;
    #pragma unroll
    for (int j = 0; j < 8; ++j) acc[j] = bias[wo + j + hi * 8];
    const float* Wrow = W + (size_t)(wo + col) * 512;
    #pragma unroll 1
    for (int k = 0; k < 16; ++k) {             // K = 512, 32 per WMMA
      const v16bf a = pack_a(Wrow, k * 32 + hi * 16);
      const v16bf bx = *(const v16bf*)&xs[(lt * 16 + col) * CPAD + k * 32 + hi * 16];
      acc = WMMA_BF16(a, bx, acc);
    }
    bf16* Yo = Y + (size_t)(b * 384 + o0 + hi * 8) * 4096 + l0 + lt * 16 + col;
    #pragma unroll
    for (int j = 0; j < 8; ++j) Yo[(size_t)j * 4096] = (bf16)acc[j];
  }
}

// ---------------------------------------------------------------------------
// Kernel 2: repack theta to [b][h][l][c8]; 2x2-maxpool phi to [b][h][d][c8]
// and g to [b][h][e][d] (all bf16, layouts chosen for WMMA fragment loads).
// ---------------------------------------------------------------------------
__global__ __launch_bounds__(256) void k_pool(
    const bf16* __restrict__ Y,
    bf16* __restrict__ thetaT, bf16* __restrict__ phiT, bf16* __restrict__ gT)
{
  const int idx = blockIdx.x * 256 + threadIdx.x;
  const int NT = 4 * 8 * 4096 * 8;     // 1048576
  const int NP = 4 * 8 * 1024 * 8;     //  262144
  const int NG = 4 * 8 * 32 * 1024;    // 1048576
  if (idx < NT) {
    const int c = idx & 7, l = (idx >> 3) & 4095, h = (idx >> 15) & 7, b = idx >> 18;
    thetaT[idx] = Y[(size_t)(b * 384 + h * 8 + c) * 4096 + l];
  } else if (idx < NT + NP) {
    const int k = idx - NT;
    const int c = k & 7, d = (k >> 3) & 1023, h = (k >> 13) & 7, b = k >> 16;
    const int l00 = (d >> 5) * 128 + (d & 31) * 2;
    const bf16* Yo = Y + (size_t)(b * 384 + 64 + h * 8 + c) * 4096;
    const float m = fmaxf(fmaxf((float)Yo[l00], (float)Yo[l00 + 1]),
                          fmaxf((float)Yo[l00 + 64], (float)Yo[l00 + 65]));
    phiT[k] = (bf16)m;
  } else if (idx < NT + NP + NG) {
    const int k = idx - NT - NP;
    const int d = k & 1023, e = (k >> 10) & 31, h = (k >> 15) & 7, b = k >> 18;
    const int l00 = (d >> 5) * 128 + (d & 31) * 2;
    const bf16* Yo = Y + (size_t)(b * 384 + 128 + h * 32 + e) * 4096;
    const float m = fmaxf(fmaxf((float)Yo[l00], (float)Yo[l00 + 1]),
                          fmaxf((float)Yo[l00 + 64], (float)Yo[l00 + 65]));
    gT[k] = (bf16)m;
  }
}

// ---------------------------------------------------------------------------
// Kernel 3: fused masked attention.  Per wave: one 16-row l-tile of one (b,h).
// Scores via WMMA (K=8 zero-padded to 32), analytic grid mask, exp (masked ->
// exact 0), probabilities staged in LDS to transpose into the A layout, then
// P.g via WMMA with an extra all-ones B tile producing the softmax row sums.
// Mask-aware chunk skipping: block-pattern heads touch exactly 1 of 32
// d-chunks; strided heads only chunks on the valid side of the diagonal.
// ---------------------------------------------------------------------------
__global__ __launch_bounds__(256) void k_attn(
    const bf16* __restrict__ thetaT,
    const bf16* __restrict__ phiT,
    const bf16* __restrict__ gT,
    bf16* __restrict__ agT)
{
  constexpr int PR = 40;                        // 80B row stride: aligned, conflict-free
  __shared__ __align__(16) bf16 ps[8 * 16 * PR];
  const int bh = blockIdx.x >> 5;               // (b,h)
  const int b = bh >> 3, h = bh & 7;
  const int wv = threadIdx.x >> 5, lane = threadIdx.x & 31;
  const int lt = (blockIdx.x & 31) * 8 + wv;    // l-tile 0..255
  const int l0 = lt * 16;
  const int blk = (l0 >> 6) >> 1;               // i>>5 (constant over tile)
  const int col = lane & 15, hi = lane >> 4;
  bf16* myps = &ps[wv * 16 * PR];

  const bf16* thb = thetaT + ((size_t)bh * 4096 + l0) * 8;
  const bf16* phb = phiT + (size_t)bh * 1024 * 8;
  const bf16* gb  = gT + (size_t)bh * 32 * 1024;

  BFrag aT; zero_frag(aT);                      // theta A: K=0..7 valid, rest 0
  if (lane < 16) aT.q[0] = *(const uint4*)(thb + (size_t)lane * 8);

  v16bf ones;
  #pragma unroll
  for (int q = 0; q < 16; ++q) ones[q] = (bf16)1.0f;

  int c_lo, c_hi;
  if ((h & 1) == 0) { c_lo = blk; c_hi = blk + 1; }   // block pattern: 1 chunk
  else if (h < 4)   { c_lo = blk; c_hi = 32; }        // strided, j >= i
  else              { c_lo = 0;   c_hi = blk + 1; }   // strided, j <= i

  v8f O0 = {}, O1 = {}, Os = {};

  for (int ch = c_lo; ch < c_hi; ++ch) {
    const int d0 = ch * 32;
    BFrag pb0, pb1; zero_frag(pb0); zero_frag(pb1);   // phi B: K=0..7 valid
    if (lane < 16) {
      pb0.q[0] = *(const uint4*)(phb + (size_t)(d0 + col) * 8);
      pb1.q[0] = *(const uint4*)(phb + (size_t)(d0 + 16 + col) * 8);
    }
    v8f S0 = {}, S1 = {};
    S0 = WMMA_BF16(aT.v, pb0.v, S0);
    S1 = WMMA_BF16(aT.v, pb1.v, S1);

    #pragma unroll
    for (int j = 0; j < 8; ++j) {
      const int lrow = j + hi * 8;
      const int i_ = blk * 32 + (((l0 & 63) + lrow) >> 1);
      {
        const int jd = d0 + col;
        const bool dir = (h < 4) ? (jd >= i_) : (jd <= i_);
        const bool pat = (h & 1) ? (((i_ - jd) & 31) == 0) : ((i_ >> 5) == (jd >> 5));
        const float p = (dir && pat) ? __expf(S0[j]) : 0.0f;
        myps[lrow * PR + col] = (bf16)p;
      }
      {
        const int jd = d0 + 16 + col;
        const bool dir = (h < 4) ? (jd >= i_) : (jd <= i_);
        const bool pat = (h & 1) ? (((i_ - jd) & 31) == 0) : ((i_ >> 5) == (jd >> 5));
        const float p = (dir && pat) ? __expf(S1[j]) : 0.0f;
        myps[lrow * PR + 16 + col] = (bf16)p;
      }
    }
    asm volatile("s_wait_dscnt 0x0" ::: "memory");    // LDS store -> load, same wave

    BFrag pa;                                          // P as A (16x32, transposed)
    {
      const bf16* base = myps + col * PR + hi * 16;
      pa.q[0] = *(const uint4*)(base);
      pa.q[1] = *(const uint4*)(base + 8);
    }
    BFrag g0, g1;                                      // g as B: K=d fully valid
    {
      const bf16* p0 = gb + (size_t)col * 1024 + d0 + hi * 16;
      g0.q[0] = *(const uint4*)(p0);
      g0.q[1] = *(const uint4*)(p0 + 8);
      const bf16* p1 = gb + (size_t)(16 + col) * 1024 + d0 + hi * 16;
      g1.q[0] = *(const uint4*)(p1);
      g1.q[1] = *(const uint4*)(p1 + 8);
    }
    O0 = WMMA_BF16(pa.v, g0.v, O0);
    O1 = WMMA_BF16(pa.v, g1.v, O1);
    Os = WMMA_BF16(pa.v, ones, Os);                    // row sums, all lanes
  }

  bf16* outb = agT + ((size_t)b * 4096 + l0) * 256 + h * 32;  // [b][l][cg]
  #pragma unroll
  for (int j = 0; j < 8; ++j) {
    const float inv = 1.0f / Os[j];                    // > 0: diagonal always unmasked
    const int lrow = j + hi * 8;
    outb[(size_t)lrow * 256 + col]      = (bf16)(O0[j] * inv);
    outb[(size_t)lrow * 256 + 16 + col] = (bf16)(O1[j] * inv);
  }
}

// ---------------------------------------------------------------------------
// Kernel 4: out = x + sigma * (Wo . attn_g + bo).  LDS-tiled WMMA GEMM,
// K = 256 (attn_g channels), epilogue reads x / writes fp32 coalesced.
// ---------------------------------------------------------------------------
__global__ __launch_bounds__(256) void k_out(
    const bf16* __restrict__ agT,
    const float* __restrict__ Wo, const float* __restrict__ bo,
    const float* __restrict__ x, const float* __restrict__ sigma,
    float* __restrict__ out)
{
  constexpr int KP = 272;
  __shared__ __align__(16) bf16 ag[64 * KP];
  const int b  = blockIdx.x >> 6;
  const int l0 = (blockIdx.x & 63) * 64;
  const int tid = threadIdx.x;
  const bf16* src = agT + ((size_t)b * 4096 + l0) * 256;
  for (int idx = tid; idx < 64 * 128; idx += 256) {
    const int li = idx >> 7, c2 = (idx & 127) * 2;
    *(uint32_t*)&ag[li * KP + c2] = *(const uint32_t*)&src[li * 256 + c2];
  }
  __syncthreads();
  const float sg = sigma[0];
  const int wv = tid >> 5, lane = tid & 31;
  const int col = lane & 15, hi = lane >> 4;
  for (int t = wv; t < 128; t += 8) {          // 32 o-tiles * 4 l-tiles
    const int ot = t >> 2, lt = t & 3;
    const int o0 = ot * 16;
    const float* Wrow = Wo + (size_t)(o0 + col) * 256;
    v8f acc = {};
    #pragma unroll 1
    for (int k = 0; k < 8; ++k) {
      const v16bf a = pack_a(Wrow, k * 32 + hi * 16);
      const v16bf bx = *(const v16bf*)&ag[(lt * 16 + col) * KP + k * 32 + hi * 16];
      acc = WMMA_BF16(a, bx, acc);
    }
    #pragma unroll
    for (int j = 0; j < 8; ++j) {
      const int o = o0 + j + hi * 8;
      const size_t xi = ((size_t)(b * 512 + o)) * 4096 + l0 + lt * 16 + col;
      out[xi] = x[xi] + sg * (acc[j] + bo[o]);
    }
  }
}

// ---------------------------------------------------------------------------
extern "C" void kernel_launch(void* const* d_in, const int* in_sizes, int n_in,
                              void* d_out, int out_size, void* d_ws, size_t ws_size,
                              hipStream_t stream) {
  (void)in_sizes; (void)n_in; (void)out_size; (void)ws_size;
  const float* x     = (const float*)d_in[0];
  const float* Wt    = (const float*)d_in[1];
  const float* bt    = (const float*)d_in[2];
  const float* Wp    = (const float*)d_in[3];
  const float* bp    = (const float*)d_in[4];
  const float* Wg    = (const float*)d_in[5];
  const float* bg    = (const float*)d_in[6];
  const float* Wo    = (const float*)d_in[7];
  const float* bo    = (const float*)d_in[8];
  const float* sigma = (const float*)d_in[9];
  // d_in[10] (masks, 128MB) deliberately unused: regenerated analytically.

  char* ws = (char*)d_ws;
  bf16* Y      = (bf16*)(ws);                                        // 12,582,912 B
  bf16* thetaT = (bf16*)(ws + 12582912);                             //  2,097,152 B
  bf16* phiT   = (bf16*)(ws + 12582912 + 2097152);                   //    524,288 B
  bf16* gT     = (bf16*)(ws + 12582912 + 2097152 + 524288);          //  2,097,152 B
  bf16* agT    = (bf16*)(ws + 12582912 + 2097152 + 524288 + 2097152);//  8,388,608 B

  k_proj<<<512, 256, 0, stream>>>(x, Wt, bt, Wp, bp, Wg, bg, Y);
  k_pool<<<9216, 256, 0, stream>>>(Y, thetaT, phiT, gT);
  k_attn<<<1024, 256, 0, stream>>>(thetaT, phiT, gT, agT);
  k_out<<<256, 256, 0, stream>>>(agT, Wo, bo, x, sigma, (float*)d_out);
}
